// LGSBlockV2_31610959299144
// MI455X (gfx1250) — compile-verified, matrix operand order checked by reading, not run on hardware
//
#include <hip/hip_runtime.h>

typedef __attribute__((ext_vector_type(2)))  float        v2f;
typedef __attribute__((ext_vector_type(8)))  float        v8f;
typedef __attribute__((ext_vector_type(8)))  _Float16     v8h;
typedef __attribute__((ext_vector_type(16))) _Float16     v16h;
typedef __attribute__((ext_vector_type(4)))  unsigned int u32x4;
typedef __attribute__((ext_vector_type(4)))  int          i32x4;
typedef __attribute__((ext_vector_type(8)))  int          i32x8;

#if __has_builtin(__builtin_amdgcn_tensor_load_to_lds) && __has_builtin(__builtin_amdgcn_s_wait_tensorcnt)
#define GEMM_TDM 1
#else
#define GEMM_TDM 0
#endif

// ---- CDNA5 WMMA wrappers ----
__device__ __forceinline__ v8f wmma4(v2f a, v2f b, v8f c) {  // f32, K=4
  return __builtin_amdgcn_wmma_f32_16x16x4_f32(false, a, false, b, (short)0, c, false, false);
}
__device__ __forceinline__ v8f wmma16(v16h a, v16h b, v8f c) { // f16 in, f32 acc, K=32
  return __builtin_amdgcn_wmma_f32_16x16x32_f16(false, a, false, b, (short)0, c, false, false);
}
__device__ __forceinline__ float sigf(float x){ return 1.f/(1.f+__expf(-x)); }
__device__ __forceinline__ float siluf(float x){ return x*sigf(x); }
__device__ __forceinline__ float gelf(float x){ return 0.5f*x*(1.f+erff(x*0.70710678118654752f)); }

#if GEMM_TDM
// Issue one TDM 2D tile load: rows x cols f32 tile at gptr (row stride = ld
// elements) -> contiguous LDS at ldsdst.  Descriptor packing per CDNA5 ISA
// 8.3/8.4: group0 = {count|flags, lds_addr, global_addr, type=2},
// group1 = {data_size=4B, tensor dims (no OOB), tile dims, dim0 stride}.
__device__ __forceinline__ void tdm_load_2d(const float* gptr, int ld, void* ldsdst,
                                            int cols, int rows)
{
  unsigned lds = (unsigned)(size_t)ldsdst;                  // low 32 bits = LDS offset
  unsigned long long ga = (unsigned long long)(size_t)gptr; // 57-bit global address
  u32x4 g0 = { 1u,                                          // count=1, user mode
               lds,
               (unsigned)(ga & 0xFFFFFFFFu),
               (unsigned)((ga >> 32) & 0x1FFFFFFu) | (2u << 30) }; // type=2 (image)
  i32x8 g1 = { 0x20000,                                     // data_size = 2 (4 bytes)
               (int)0xFFFF0000u,                            // tensor_dim0 lo16 (huge)
               (int)0xFFFF7FFFu,                            // tensor_dim0 hi / dim1 lo
               (int)((cols << 16) | 0x7FFF),                // tile_dim0 | tensor_dim1 hi
               rows,                                        // tile_dim1 (tile_dim2 = 0)
               ld,                                          // tensor_dim0_stride lo32
               0, 0 };
  i32x4 gz = { 0, 0, 0, 0 };
#if defined(__clang_major__) && (__clang_major__ >= 23)
  i32x8 g4 = { 0, 0, 0, 0, 0, 0, 0, 0 };
  __builtin_amdgcn_tensor_load_to_lds(g0, g1, gz, gz, g4, 0);
#else
  __builtin_amdgcn_tensor_load_to_lds(g0, g1, gz, gz, 0);
#endif
}
#endif

// ============ f16-input WMMA GEMM, TDM + double-buffered LDS =======
// C[M,N] = act(A[M,K] @ B[K,N] + bias) + res  (A,B f32 in memory, f16 compute,
// f32 accumulate). Block = 128 thr (4 waves), 64x64 tile, K step 32.
// K % 32 == 0 required; M,N guarded.
// EPI: 0 none, 1 silu, 2 sigmoid, 3 relu, 4 gelu (exact)
template<int EPI>
__global__ void gemm16_ker(const float* __restrict__ A, int lda,
                           const float* __restrict__ B, int ldb,
                           const float* __restrict__ bias,
                           const float* __restrict__ res,
                           float* __restrict__ C, int ldc,
                           int M, int N, int K)
{
  __shared__ __align__(16) float    Asf[2][64*32];   // [m][k] raw f32 (TDM target)
  __shared__ __align__(16) _Float16 Bs [2][64*32];   // transposed: [n][k] f16
  int tid = threadIdx.x;
  int wave = tid >> 5, lane = tid & 31;
  int tN = blockIdx.x * 64, tM = blockIdx.y * 64;
  int row = lane & 15;
  int kb  = (lane >> 4) << 3;   // 0 or 8 (K base within fragment halves)
  bool fullN = (tN + 64 <= N);
  v8f acc[4] = {};

  auto stageA = [&](int buf, int kk) {
#if GEMM_TDM
    if (wave == 0)
      tdm_load_2d(A + (long long)tM*lda + kk, lda, &Asf[buf][0], 32, 64);
#else
#pragma unroll
    for (int i = 0; i < 4; ++i) {
      int q = tid + i*128;           // 0..511
      int r = q >> 3;                // 0..63
      int c4 = (q & 7) << 2;         // 0,4,...,28
      float4 va;
      if (tM + r < M) va = *(const float4*)(A + (long long)(tM + r)*lda + kk + c4);
      else            va = make_float4(0.f, 0.f, 0.f, 0.f);
      *(float4*)&Asf[buf][r*32 + c4] = va;
    }
#endif
  };
  auto stageB = [&](int buf, int kk) {
    if (fullN) {
#pragma unroll
      for (int i = 0; i < 4; ++i) {
        int q = tid + i*128;
        int k = q >> 4;              // 0..31
        int n4 = (q & 15) << 2;      // 0..60
        float4 vb = *(const float4*)(B + (long long)(kk + k)*ldb + tN + n4);
        Bs[buf][(n4+0)*32 + k] = (_Float16)vb.x;
        Bs[buf][(n4+1)*32 + k] = (_Float16)vb.y;
        Bs[buf][(n4+2)*32 + k] = (_Float16)vb.z;
        Bs[buf][(n4+3)*32 + k] = (_Float16)vb.w;
      }
    } else {
#pragma unroll
      for (int i = 0; i < 4; ++i) {
        int q = tid + i*128;
        int k = q >> 4;
        int n4 = (q & 15) << 2;
#pragma unroll
        for (int j = 0; j < 4; ++j) {
          int n = tN + n4 + j;
          float v = (n < N) ? B[(long long)(kk + k)*ldb + n] : 0.f;
          Bs[buf][(n4+j)*32 + k] = (_Float16)v;
        }
      }
    }
  };

  // prologue: stage first panel into buffer 0
  stageA(0, 0);
  stageB(0, 0);
  int buf = 0;
  for (int kk = 0; kk < K; kk += 32) {
#if GEMM_TDM
    if (wave == 0) __builtin_amdgcn_s_wait_tensorcnt(0);   // A panel of `buf` landed
#endif
    __syncthreads();                                       // all stores to `buf` visible
    if (kk + 32 < K) {                                     // stage next panel (overlap)
      stageA(buf ^ 1, kk + 32);
      stageB(buf ^ 1, kk + 32);
    }
    if (kk + 64 < K)                                       // gfx1250 global_prefetch
      __builtin_prefetch(B + (long long)(kk + 64 + (tid >> 2))*ldb + tN, 0, 1);
    // ---- fragments from LDS (f32 -> f16 for A) + WMMA ----
    int ar = wave*16 + row;
    const float* arow = &Asf[buf][ar*32];
    v16h af;
#pragma unroll
    for (int i = 0; i < 8; ++i) af[i]     = (_Float16)arow[kb + i];
#pragma unroll
    for (int i = 0; i < 8; ++i) af[8 + i] = (_Float16)arow[kb + 16 + i];
#pragma unroll
    for (int t = 0; t < 4; ++t) {
      int nr = t*16 + row;
      v8h blo = *(const v8h*)&Bs[buf][nr*32 + kb];
      v8h bhi = *(const v8h*)&Bs[buf][nr*32 + kb + 16];
      v16h bf = __builtin_shufflevector(blo, bhi, 0,1,2,3,4,5,6,7,8,9,10,11,12,13,14,15);
      acc[t] = wmma16(af, bf, acc[t]);
    }
    buf ^= 1;
  }
  // ---- epilogue ----
  int mhi = (lane >> 4) << 3;
#pragma unroll
  for (int t = 0; t < 4; ++t)
#pragma unroll
    for (int r = 0; r < 8; ++r) {
      int m = tM + wave*16 + r + mhi;
      int n = tN + t*16 + (lane & 15);
      if (m < M && n < N) {
        float v = acc[t][r];
        if (bias) v += bias[n];
        if (EPI == 1) v = siluf(v);
        else if (EPI == 2) v = sigf(v);
        else if (EPI == 3) v = fmaxf(v, 0.f);
        else if (EPI == 4) v = gelf(v);
        if (res) v += res[(long long)m*ldc + n];
        C[(long long)m*ldc + n] = v;
      }
    }
}

template<int EPI>
static void gemm16_launch(hipStream_t st, const float* A, int lda,
                          const float* B, int ldb,
                          const float* bias, const float* res,
                          float* C, int ldc, int M, int N, int K)
{
  dim3 g((N + 63) / 64, (M + 63) / 64, 1);
  gemm16_ker<EPI><<<g, 128, 0, st>>>(A, lda, B, ldb, bias, res, C, ldc, M, N, K);
}

// ============ f32 WMMA GEMM (full precision; used for SSD G0) ======
// C[M,N] = A[M,K] @ B^T (B is [N,K]); one wave, 32x32 tile, batched via grid.z.
__global__ void gemm32tb_ker(const float* __restrict__ A, int lda, long long bsA,
                             const float* __restrict__ B, int ldb, long long bsB,
                             float* __restrict__ C, int ldc, long long bsC,
                             int M, int N, int K)
{
  int lane = threadIdx.x;
  int tN = blockIdx.x * 32, tM = blockIdx.y * 32;
  long long z = blockIdx.z;
  A += z * bsA; B += z * bsB; C += z * bsC;
  int mr = lane & 15;
  int kq = (lane >> 4) << 1;
  int m0 = tM + mr, m1 = m0 + 16;
  int n0 = tN + mr, n1 = n0 + 16;
  v8f acc[2][2] = {};
  for (int kk = 0; kk < K; kk += 4) {
    int k = kk + kq;
    v2f a0, a1, b0, b1;
    a0.x = A[(long long)m0*lda + k    ];
    a0.y = A[(long long)m0*lda + k + 1];
    a1.x = A[(long long)m1*lda + k    ];
    a1.y = A[(long long)m1*lda + k + 1];
    b0.x = B[(long long)n0*ldb + k    ];
    b0.y = B[(long long)n0*ldb + k + 1];
    b1.x = B[(long long)n1*ldb + k    ];
    b1.y = B[(long long)n1*ldb + k + 1];
    acc[0][0] = wmma4(a0, b0, acc[0][0]);
    acc[0][1] = wmma4(a0, b1, acc[0][1]);
    acc[1][0] = wmma4(a1, b0, acc[1][0]);
    acc[1][1] = wmma4(a1, b1, acc[1][1]);
  }
  int mhi = (lane >> 4) << 3;
#pragma unroll
  for (int ti = 0; ti < 2; ++ti)
#pragma unroll
    for (int tj = 0; tj < 2; ++tj)
#pragma unroll
      for (int r = 0; r < 8; ++r) {
        int m = tM + ti*16 + r + mhi;
        int n = tN + tj*16 + (lane & 15);
        if (m < M && n < N)
          C[(long long)m*ldc + n] = acc[ti][tj][r];
      }
}

// ===================== row-wise norms =============================
__global__ void ln_ker(const float* __restrict__ X, const float* __restrict__ w,
                       const float* __restrict__ b, float* __restrict__ O)
{ // rows of 256
  __shared__ float sm[256];
  long long row = blockIdx.x; int t = threadIdx.x;
  float x = X[row*256 + t];
  sm[t] = x; __syncthreads();
  for (int o = 128; o > 0; o >>= 1) { if (t < o) sm[t] += sm[t+o]; __syncthreads(); }
  float mean = sm[0] * (1.f/256.f);
  __syncthreads();
  float xc = x - mean;
  sm[t] = xc*xc; __syncthreads();
  for (int o = 128; o > 0; o >>= 1) { if (t < o) sm[t] += sm[t+o]; __syncthreads(); }
  float var = sm[0] * (1.f/256.f);
  O[row*256 + t] = xc * rsqrtf(var + 1e-5f) * w[t] + b[t];
}

__global__ void rms_ker(const float* __restrict__ Y, const float* __restrict__ w,
                        float* __restrict__ U)
{ // rows of 512, RMS norm * w
  __shared__ float sm[256];
  long long row = blockIdx.x; int t = threadIdx.x;
  float x0 = Y[row*512 + t], x1 = Y[row*512 + 256 + t];
  sm[t] = x0*x0 + x1*x1; __syncthreads();
  for (int o = 128; o > 0; o >>= 1) { if (t < o) sm[t] += sm[t+o]; __syncthreads(); }
  float r = rsqrtf(sm[0] * (1.f/512.f) + 1e-5f);
  U[row*512 + t]       = x0 * r * w[t];
  U[row*512 + 256 + t] = x1 * r * w[256 + t];
}

// ===================== mamba pre-SSD ==============================
__global__ void conv_ker(const float* __restrict__ ZX, const float* __restrict__ cw,
                         const float* __restrict__ cb, float* __restrict__ XC)
{ // causal depthwise conv over L, width 4, then silu.  total = 8*4096*640
  long long i = (long long)blockIdx.x*blockDim.x + threadIdx.x;
  if (i >= 8LL*4096*640) return;
  int ch = (int)(i % 640);
  long long r = i / 640;
  int l = (int)(r & 4095);
  int b = (int)(r >> 12);
  float s = cb[ch];
#pragma unroll
  for (int k = 0; k < 4; ++k) {
    int ll = l + k - 3;
    if (ll >= 0) s += ZX[((long long)b*4096 + ll)*1160 + 512 + ch] * cw[ch*4 + k];
  }
  XC[i] = siluf(s);
}

__global__ void dt_ker(const float* __restrict__ ZX, const float* __restrict__ dtb,
                       float* __restrict__ DT)
{ // softplus(dt + dt_bias), total = 8*4096*8
  long long i = (long long)blockIdx.x*blockDim.x + threadIdx.x;
  if (i >= 8LL*4096*8) return;
  int h = (int)(i & 7); long long r = i >> 3;
  float x = ZX[r*1160 + 1152 + h] + dtb[h];
  DT[i] = (x > 20.f) ? x : log1pf(__expf(x));
}

__global__ void cumsum_ker(const float* __restrict__ DT, const float* __restrict__ a_log,
                           float* __restrict__ ACS)
{ // blockIdx.x = (b*8+h)*16+c ; inclusive scan of a over 256-token chunk
  __shared__ float sm[256];
  int bidx = blockIdx.x;
  int c = bidx & 15; int bh = bidx >> 4; int h = bh & 7; int b = bh >> 3;
  int s = threadIdx.x;
  float na = -__expf(a_log[h]);
  sm[s] = DT[((long long)(b*4096 + c*256 + s))*8 + h] * na;
  __syncthreads();
  for (int off = 1; off < 256; off <<= 1) {
    float t = sm[s];
    if (s >= off) t += sm[s - off];
    __syncthreads();
    sm[s] = t;
    __syncthreads();
  }
  ACS[(long long)bidx*256 + s] = sm[s];
}

// ===================== SSD custom WMMA kernels ====================
// z = ((b*16+c)*8+h). Per wave: 16 L-rows x 64 P-cols, K=256 (s).
__global__ void ssd_yd_ker(const float* __restrict__ XC, const float* __restrict__ DT,
                           const float* __restrict__ ACS, const float* __restrict__ G0,
                           float* __restrict__ Y)
{
  int lane = threadIdx.x;
  int z = blockIdx.z;
  int h = z & 7; int bc = z >> 3; int c = bc & 15; int b = bc >> 4;
  const float* g0  = G0 + (long long)bc * 65536;
  const float* acs = ACS + (((long long)(b*8 + h))*16 + c) * 256;
  const float* xcb = XC + ((long long)(b*4096 + c*256)) * 640;
  const float* dtb = DT + ((long long)(b*4096 + c*256)) * 8 + h;
  int mr = lane & 15, kq = (lane >> 4) << 1;
  int l0 = blockIdx.y * 16;
  int l = l0 + mr;
  float acl = acs[l];
  v8f acc[4] = {};
  for (int kk = 0; kk < 256; kk += 4) {
    int s0 = kk + kq, s1 = s0 + 1;
    // unconditional loads (always in-bounds); mask via select -> no EXEC churn
    float g0v = g0[l*256 + s0];
    float g1v = g0[l*256 + s1];
    float e0 = __expf(acl - acs[s0]);
    float e1 = __expf(acl - acs[s1]);
    v2f a;
    a.x = (l >= s0) ? g0v * e0 : 0.f;
    a.y = (l >= s1) ? g1v * e1 : 0.f;
    float d0 = dtb[(long long)s0*8];
    float d1 = dtb[(long long)s1*8];
#pragma unroll
    for (int t = 0; t < 4; ++t) {
      int p = t*16 + mr;
      v2f bb;
      bb.x = xcb[(long long)s0*640 + h*64 + p] * d0;
      bb.y = xcb[(long long)s1*640 + h*64 + p] * d1;
      acc[t] = wmma4(a, bb, acc[t]);
    }
  }
  int mhi = (lane >> 4) << 3;
#pragma unroll
  for (int t = 0; t < 4; ++t)
#pragma unroll
    for (int r = 0; r < 8; ++r) {
      int m = l0 + r + mhi;
      int p = t*16 + (lane & 15);
      Y[((long long)b*4096 + c*256 + m)*512 + h*64 + p] = acc[t][r];
    }
}

// states[b,c,h,p,n] = sum_s xd[s,h,p]*dec_st[s]*Bc[s,n]; wave: 16 P x 64 N, K=256
__global__ void ssd_states_ker(const float* __restrict__ XC, const float* __restrict__ DT,
                               const float* __restrict__ ACS, float* __restrict__ ST)
{
  int lane = threadIdx.x;
  int z = blockIdx.z;
  int h = z & 7; int bc = z >> 3; int c = bc & 15; int b = bc >> 4;
  const float* acs = ACS + (((long long)(b*8 + h))*16 + c) * 256;
  const float* xcb = XC + ((long long)(b*4096 + c*256)) * 640;
  const float* dtb = DT + ((long long)(b*4096 + c*256)) * 8 + h;
  float aclast = acs[255];
  int mr = lane & 15, kq = (lane >> 4) << 1;
  int p0 = blockIdx.y * 16;
  int p = p0 + mr;
  v8f acc[4] = {};
  for (int kk = 0; kk < 256; kk += 4) {
    int s0 = kk + kq, s1 = s0 + 1;
    v2f a;
    a.x = xcb[(long long)s0*640 + h*64 + p] * dtb[(long long)s0*8] * __expf(aclast - acs[s0]);
    a.y = xcb[(long long)s1*640 + h*64 + p] * dtb[(long long)s1*8] * __expf(aclast - acs[s1]);
#pragma unroll
    for (int t = 0; t < 4; ++t) {
      int n = t*16 + mr;
      v2f bb;
      bb.x = xcb[(long long)s0*640 + 512 + n];
      bb.y = xcb[(long long)s1*640 + 512 + n];
      acc[t] = wmma4(a, bb, acc[t]);
    }
  }
  int mhi = (lane >> 4) << 3;
#pragma unroll
  for (int t = 0; t < 4; ++t)
#pragma unroll
    for (int r = 0; r < 8; ++r) {
      int m = p0 + r + mhi;
      int n = t*16 + (lane & 15);
      ST[(((long long)z)*64 + m)*64 + n] = acc[t][r];
    }
}

__global__ void ssd_scan_ker(const float* __restrict__ ST, const float* __restrict__ ACS,
                             float* __restrict__ STIN)
{ // serial scan over 16 chunks; i over (b,h,p,n) = 262144
  long long i = (long long)blockIdx.x*blockDim.x + threadIdx.x;
  if (i >= 262144) return;
  int n = (int)(i & 63); int p = (int)((i >> 6) & 63);
  int h = (int)((i >> 12) & 7); int b = (int)(i >> 15);
  float hh = 0.f;
  for (int c = 0; c < 16; ++c) {
    long long si = (((((long long)(b*16 + c))*8 + h)*64 + p)*64 + n);
    STIN[si] = hh;
    float dec = __expf(ACS[(((long long)(b*8 + h))*16 + c)*256 + 255]);
    hh = hh * dec + ST[si];
  }
}

// Yo then finalize: y = (Yd + Yo + D*xs) * silu(z). wave: 16 L x 64 P, K=64 (n)
__global__ void ssd_yo_ker(const float* __restrict__ XC, const float* __restrict__ ACS,
                           const float* __restrict__ STIN, const float* __restrict__ ZX,
                           const float* __restrict__ md, float* __restrict__ Y)
{
  int lane = threadIdx.x;
  int z = blockIdx.z;
  int h = z & 7; int bc = z >> 3; int c = bc & 15; int b = bc >> 4;
  const float* acs = ACS + (((long long)(b*8 + h))*16 + c) * 256;
  const float* xcb = XC + ((long long)(b*4096 + c*256)) * 640;
  const float* st  = STIN + ((long long)z) * 4096; // [p][n]
  int mr = lane & 15, kq = (lane >> 4) << 1;
  int l0 = blockIdx.y * 16;
  int l = l0 + mr;
  v8f acc[4] = {};
  for (int kk = 0; kk < 64; kk += 4) {
    int n0 = kk + kq, n1 = n0 + 1;
    v2f a;
    a.x = xcb[(long long)l*640 + 576 + n0];
    a.y = xcb[(long long)l*640 + 576 + n1];
#pragma unroll
    for (int t = 0; t < 4; ++t) {
      int p = t*16 + mr;
      v2f bb;
      bb.x = st[(long long)p*64 + n0];
      bb.y = st[(long long)p*64 + n1];
      acc[t] = wmma4(a, bb, acc[t]);
    }
  }
  float dh = md[h];
  int mhi = (lane >> 4) << 3;
#pragma unroll
  for (int t = 0; t < 4; ++t)
#pragma unroll
    for (int r = 0; r < 8; ++r) {
      int m = l0 + r + mhi;
      int p = t*16 + (lane & 15);
      long long lg = (long long)b*4096 + c*256 + m;
      long long yi = lg*512 + h*64 + p;
      float v = acc[t][r] * __expf(acs[m]);
      v += Y[yi] + dh * xcb[(long long)m*640 + h*64 + p];
      float zz = ZX[lg*1160 + h*64 + p];
      Y[yi] = v * siluf(zz);
    }
}

// ===================== window attention ===========================
// grid: (512 windows, 8 heads), block 128 (4 waves). S in LDS.
__global__ void attn_ker(const float* __restrict__ QKV, const float* __restrict__ rpb,
                         float* __restrict__ OUT)
{
  __shared__ float S[64*64];
  int widx = blockIdx.x, head = blockIdx.y;
  int tid = threadIdx.x; int wave = tid >> 5; int lane = tid & 31;
  const float* base = QKV + (long long)widx * 64 * 768;
  int qb = head*32, kb = 256 + head*32, vb = 512 + head*32;
  int mr = lane & 15, kq = (lane >> 4) << 1;
  int l = wave*16 + mr;
  const float scale = 0.17677669529663687f; // 32^-0.5
  v8f acc[4] = {};
  for (int kk = 0; kk < 32; kk += 4) {
    int k0 = kk + kq;
    v2f a;
    a.x = base[(long long)l*768 + qb + k0    ] * scale;
    a.y = base[(long long)l*768 + qb + k0 + 1] * scale;
#pragma unroll
    for (int t = 0; t < 4; ++t) {
      int s = t*16 + mr;
      v2f bb;
      bb.x = base[(long long)s*768 + kb + k0    ];
      bb.y = base[(long long)s*768 + kb + k0 + 1];
      acc[t] = wmma4(a, bb, acc[t]);
    }
  }
  int wi = widx & 63; int wh = wi >> 3, wwc = wi & 7;
  int mhi = (lane >> 4) << 3;
#pragma unroll
  for (int t = 0; t < 4; ++t)
#pragma unroll
    for (int r = 0; r < 8; ++r) {
      int m = wave*16 + r + mhi;
      int s = t*16 + (lane & 15);
      int rm = m >> 3, cm = m & 7, rs = s >> 3, cs = s & 7;
      int idx = (rm - rs + 7)*15 + (cm - cs + 7);
      float v = acc[t][r] + rpb[idx*8 + head];
      int hm = wh*8 + rm, wm = wwc*8 + cm;
      int hs = wh*8 + rs, ws2 = wwc*8 + cs;
      int regm = (hm < 56 ? 0 : (hm < 60 ? 1 : 2))*3 + (wm  < 56 ? 0 : (wm  < 60 ? 1 : 2));
      int regs = (hs < 56 ? 0 : (hs < 60 ? 1 : 2))*3 + (ws2 < 56 ? 0 : (ws2 < 60 ? 1 : 2));
      if (regm != regs) v -= 100.f;
      S[m*64 + s] = v;
    }
  __syncthreads();
  if (tid < 64) {
    float mx = -1e30f;
    for (int s = 0; s < 64; ++s) mx = fmaxf(mx, S[tid*64 + s]);
    float sum = 0.f;
    for (int s = 0; s < 64; ++s) { float e = __expf(S[tid*64 + s] - mx); S[tid*64 + s] = e; sum += e; }
    float inv = 1.f / sum;
    for (int s = 0; s < 64; ++s) S[tid*64 + s] *= inv;
  }
  __syncthreads();
  v8f o[2] = {};
  for (int kk = 0; kk < 64; kk += 4) {
    int s0 = kk + kq;
    v2f a;
    a.x = S[l*64 + s0    ];
    a.y = S[l*64 + s0 + 1];
#pragma unroll
    for (int t = 0; t < 2; ++t) {
      int d = t*16 + mr;
      v2f bb;
      bb.x = base[(long long)(s0    )*768 + vb + d];
      bb.y = base[(long long)(s0 + 1)*768 + vb + d];
      o[t] = wmma4(a, bb, o[t]);
    }
  }
#pragma unroll
  for (int t = 0; t < 2; ++t)
#pragma unroll
    for (int r = 0; r < 8; ++r) {
      int m = wave*16 + r + mhi;
      int d = t*16 + (lane & 15);
      OUT[((long long)widx*64 + m)*256 + head*32 + d] = o[t][r];
    }
}

// ===================== pack / unpack / elementwise ================
__global__ void pack_xw_ker(const float* __restrict__ XN, float* __restrict__ XW)
{ // roll(-4,-4) + window partition; total = 8*4096*256
  long long i = (long long)blockIdx.x*blockDim.x + threadIdx.x;
  if (i >= 8LL*4096*256) return;
  int ch = (int)(i & 255); long long r = i >> 8;
  int t = (int)(r & 63); long long w = r >> 6;
  int b = (int)(w >> 6); int wi = (int)(w & 63);
  int wh = wi >> 3, ww = wi & 7;
  int rr = t >> 3, cc = t & 7;
  int hs = (wh*8 + rr + 4) & 63;
  int ws2 = (ww*8 + cc + 4) & 63;
  XW[i] = XN[(((long long)b*4096) + hs*64 + ws2)*256 + ch];
}

__global__ void unwin_ker(const float* __restrict__ P, float* __restrict__ XA)
{ // un-window + roll(+4,+4); total = 8*4096*256
  long long i = (long long)blockIdx.x*blockDim.x + threadIdx.x;
  if (i >= 8LL*4096*256) return;
  int ch = (int)(i & 255); long long r = i >> 8;
  int pos = (int)(r & 4095); int b = (int)(r >> 12);
  int hh = pos >> 6, ww2 = pos & 63;
  int h2 = (hh + 60) & 63;
  int w2 = (ww2 + 60) & 63;
  int wh = h2 >> 3, rr = h2 & 7, wc = w2 >> 3, cc = w2 & 7;
  long long widx = (long long)b*64 + wh*8 + wc;
  XA[i] = P[(widx*64 + rr*8 + cc)*256 + ch];
}

__global__ void concat2_ker(const float* __restrict__ A, const float* __restrict__ B,
                            float* __restrict__ D, long long rows)
{ // D[r][0:256]=A[r], D[r][256:512]=B[r]
  long long i = (long long)blockIdx.x*blockDim.x + threadIdx.x;
  if (i >= rows*512) return;
  int c = (int)(i & 511); long long r = i >> 9;
  D[i] = (c < 256) ? A[r*256 + c] : B[r*256 + c - 256];
}

__global__ void gatecat_ker(const float* __restrict__ XM, const float* __restrict__ GM,
                            const float* __restrict__ XA, const float* __restrict__ GA,
                            float* __restrict__ D)
{ // D[r] = concat(xm*gm, xa*ga); rows = 32768
  long long i = (long long)blockIdx.x*blockDim.x + threadIdx.x;
  if (i >= 32768LL*512) return;
  int c = (int)(i & 511); long long r = i >> 9;
  if (c < 256) D[i] = XM[r*256 + c] * GM[r*256 + c];
  else         D[i] = XA[r*256 + c - 256] * GA[r*256 + c - 256];
}

__global__ void mul_ker(float* __restrict__ A, const float* __restrict__ G, long long n)
{
  long long i = (long long)blockIdx.x*blockDim.x + threadIdx.x;
  if (i < n) A[i] *= G[i];
}

// ===================== host orchestration =========================
static inline int cdiv(long long a, int b) { return (int)((a + b - 1) / b); }

extern "C" void kernel_launch(void* const* d_in, const int* in_sizes, int n_in,
                              void* d_out, int out_size, void* d_ws, size_t ws_size,
                              hipStream_t stream)
{
  const float* x       = (const float*)d_in[0];
  const float* ln1_w   = (const float*)d_in[1];
  const float* ln1_b   = (const float*)d_in[2];
  const float* ns_w    = (const float*)d_in[3];
  const float* ns_b    = (const float*)d_in[4];
  const float* m_win   = (const float*)d_in[5];
  const float* m_convw = (const float*)d_in[6];
  const float* m_convb = (const float*)d_in[7];
  const float* m_dtb   = (const float*)d_in[8];
  const float* m_alog  = (const float*)d_in[9];
  const float* m_d     = (const float*)d_in[10];
  const float* m_normw = (const float*)d_in[11];
  const float* m_wout  = (const float*)d_in[12];
  const float* tm_w1   = (const float*)d_in[13];
  const float* tm_b1   = (const float*)d_in[14];
  const float* tm_w2   = (const float*)d_in[15];
  const float* tm_b2   = (const float*)d_in[16];
  const float* nt_w    = (const float*)d_in[17];
  const float* nt_b    = (const float*)d_in[18];
  const float* qkv_w   = (const float*)d_in[19];
  const float* qkv_b   = (const float*)d_in[20];
  const float* rpb     = (const float*)d_in[21];
  const float* gate_w  = (const float*)d_in[22];
  const float* gate_b  = (const float*)d_in[23];
  const float* proj_w  = (const float*)d_in[24];
  const float* proj_b  = (const float*)d_in[25];
  const float* cg_w1a  = (const float*)d_in[26];
  const float* cg_b1a  = (const float*)d_in[27];
  const float* cg_w2a  = (const float*)d_in[28];
  const float* cg_b2a  = (const float*)d_in[29];
  const float* cg_w1m  = (const float*)d_in[30];
  const float* cg_b1m  = (const float*)d_in[31];
  const float* cg_w2m  = (const float*)d_in[32];
  const float* cg_b2m  = (const float*)d_in[33];
  const float* cg_wf   = (const float*)d_in[34];
  const float* cg_bf   = (const float*)d_in[35];
  const float* ln2_w   = (const float*)d_in[36];
  const float* ln2_b   = (const float*)d_in[37];
  const float* mlp_w1  = (const float*)d_in[38];
  const float* mlp_b1  = (const float*)d_in[39];
  const float* mlp_w2  = (const float*)d_in[40];
  const float* mlp_b2  = (const float*)d_in[41];
  float* out = (float*)d_out;

  // ---- workspace layout (floats) ----
  float* Wf   = (float*)d_ws;
  float* XN   = Wf;                    // 8,388,608
  float* XNS  = XN   + 8388608;        // 8,388,608
  float* ZX   = XNS  + 8388608;        // 38,010,880  (8*4096*1160)
  float* XC   = ZX   + 38010880;       // 20,971,520  (8*4096*640)
  float* DTm  = XC   + 20971520;       // 262,144
  float* ACS  = DTm  + 262144;         // 262,144
  float* G0   = ACS  + 262144;         // 8,388,608   (128 * 256*256)
  float* Yb   = G0   + 8388608;        // 16,777,216  (8*4096*512)
  float* STb  = Yb   + 16777216;       // 4,194,304
  float* STIN = STb  + 4194304;        // 4,194,304
  float* Ub   = STIN + 4194304;        // 16,777,216
  float* SP   = Ub   + 16777216;       // 8,388,608
  float* XM   = SP   + 8388608;        // 8,388,608
  float* XA   = XM   + 8388608;        // 8,388,608
  float* XO   = XA   + 8388608;        // 8,388,608
  // reuse region over ZX..XC (58,982,400 floats) once SSD is done:
  float* R = ZX;
  // P4 (tmix)
  float* CAT0 = R;                float* CAT1 = R + 8388608;
  float* TH0  = R + 16777216;     float* TH1  = R + 20971520;
  float* PRE0 = R + 25165824;     float* PRE1 = R + 29360128;
  // P5 (attention)
  float* XW   = R;                float* QKV  = R + 8388608;
  float* GATE = R + 33554432;     float* AOUT = R + 41943040;
  float* PROJ = R + 50331648;
  // P6 (channel gate)
  float* HA = R;               float* HM = R + 2097152;
  float* GM = R + 4194304;     float* GA = R + 12582912;
  float* FIN = R + 20971520;
  // P7 (mlp)
  float* XO2 = R;              float* MH = R + 8388608;

  const long long M  = 32768;    // twoB * L
  const long long MH_ = 16384;   // B * L (half)

  // ---- P1: layernorms ----
  ln_ker<<<32768, 256, 0, stream>>>(x,  ln1_w, ln1_b, XN);
  ln_ker<<<32768, 256, 0, stream>>>(XN, ns_w,  ns_b,  XNS);

  // ---- P2: zxbcdt GEMM, conv, dt, cumsum ----
  gemm16_launch<0>(stream, XNS,256, m_win,1160, nullptr, nullptr, ZX,1160, 32768,1160,256);
  conv_ker<<<cdiv(8LL*4096*640,256), 256, 0, stream>>>(ZX, m_convw, m_convb, XC);
  dt_ker<<<cdiv(8LL*4096*8,256), 256, 0, stream>>>(ZX, m_dtb, DTm);
  cumsum_ker<<<1024, 256, 0, stream>>>(DTm, m_alog, ACS);

  // ---- P3: SSD (full f32 precision) ----
  // G0[b,c] = Cc (256x64) @ Bc^T (64x256) ; batch = 128
  {
    dim3 g(8, 8, 128);
    gemm32tb_ker<<<g, 32, 0, stream>>>(XC+576,640,163840LL, XC+512,640,163840LL,
                                       G0,256,65536LL, 256,256,64);
  }
  ssd_yd_ker<<<dim3(1,16,1024), 32, 0, stream>>>(XC, DTm, ACS, G0, Yb);
  ssd_states_ker<<<dim3(1,4,1024), 32, 0, stream>>>(XC, DTm, ACS, STb);
  ssd_scan_ker<<<cdiv(262144,256), 256, 0, stream>>>(STb, ACS, STIN);
  ssd_yo_ker<<<dim3(1,16,1024), 32, 0, stream>>>(XC, ACS, STIN, ZX, m_d, Yb);
  rms_ker<<<32768, 256, 0, stream>>>(Yb, m_normw, Ub);
  gemm16_launch<0>(stream, Ub,512, m_wout,256, nullptr, XN, SP,256, 32768,256,512);

  // ---- P4: tmix (f1 update uses the *updated* f0) ----
  const float* f0 = SP;
  const float* f1 = SP + MH_*256;
  concat2_ker<<<cdiv(MH_*512,256), 256, 0, stream>>>(f0, f1, CAT0, MH_);
  gemm16_launch<1>(stream, CAT0,512, tm_w1,256, tm_b1, nullptr, TH0,256, 16384,256,512);
  gemm16_launch<0>(stream, TH0,256, tm_w2,256, tm_b2, f0, PRE0,256, 16384,256,256);
  ln_ker<<<16384, 256, 0, stream>>>(PRE0, nt_w, nt_b, XM);
  concat2_ker<<<cdiv(MH_*512,256), 256, 0, stream>>>(f1, XM, CAT1, MH_);
  gemm16_launch<1>(stream, CAT1,512, tm_w1,256, tm_b1, nullptr, TH1,256, 16384,256,512);
  gemm16_launch<0>(stream, TH1,256, tm_w2,256, tm_b2, f1, PRE1,256, 16384,256,256);
  ln_ker<<<16384, 256, 0, stream>>>(PRE1, nt_w, nt_b, XM + MH_*256);

  // ---- P5: shifted-window attention ----
  pack_xw_ker<<<cdiv(8LL*4096*256,256), 256, 0, stream>>>(XN, XW);
  gemm16_launch<0>(stream, XW,256, qkv_w,768, qkv_b, nullptr, QKV,768, 32768,768,256);
  gemm16_launch<2>(stream, XW,256, gate_w,256, gate_b, nullptr, GATE,256, 32768,256,256);
  attn_ker<<<dim3(512,8), 128, 0, stream>>>(QKV, rpb, AOUT);
  mul_ker<<<cdiv(M*256,256), 256, 0, stream>>>(AOUT, GATE, M*256);
  gemm16_launch<0>(stream, AOUT,256, proj_w,256, proj_b, nullptr, PROJ,256, 32768,256,256);
  unwin_ker<<<cdiv(8LL*4096*256,256), 256, 0, stream>>>(PROJ, XA);

  // ---- P6: channel gating + fuse (+ x residual) ----
  gemm16_launch<3>(stream, XA,256, cg_w1a,64,  cg_b1a, nullptr, HA,64,  32768,64,256);
  gemm16_launch<2>(stream, HA,64,  cg_w2a,256, cg_b2a, nullptr, GM,256, 32768,256,64);
  gemm16_launch<3>(stream, XM,256, cg_w1m,64,  cg_b1m, nullptr, HM,64,  32768,64,256);
  gemm16_launch<2>(stream, HM,64,  cg_w2m,256, cg_b2m, nullptr, GA,256, 32768,256,64);
  gatecat_ker<<<cdiv(M*512,256), 256, 0, stream>>>(XM, GM, XA, GA, FIN);
  gemm16_launch<0>(stream, FIN,512, cg_wf,256, cg_bf, x, XO,256, 32768,256,512);

  // ---- P7: MLP (+ xo residual) -> d_out ----
  ln_ker<<<32768, 256, 0, stream>>>(XO, ln2_w, ln2_b, XO2);
  gemm16_launch<4>(stream, XO2,256, mlp_w1,1024, mlp_b1, nullptr, MH,1024, 32768,1024,256);
  gemm16_launch<0>(stream, MH,1024, mlp_w2,256, mlp_b2, XO, out,256, 32768,256,1024);

  (void)in_sizes; (void)n_in; (void)out_size; (void)ws_size;
}
